// PNA_43705587204356
// MI455X (gfx1250) — matrix operand order repaired and checked by reading, not compiled, not run on hardware
//
#include <hip/hip_runtime.h>
#include <hip/hip_bf16.h>

// ---------------------------------------------------------------------------
// PNA forward for MI455X (gfx1250, wave32, WMMA).
//  - edge GEMM (E x 100 x 100, f16 in / f32 acc) fused with segment scatter
//  - node post-GEMM + lin GEMM via v_wmma_f32_16x16x32_f16
//  - A fragments staged in LDS (ds_load_b128), B fragments pre-packed in
//    fragment order (global_load_b128), no dynamic VGPR indexing
//  - min/max via monotone-uint atomicMin/Max; BN stats via reduced atomics
// ---------------------------------------------------------------------------

typedef _Float16 half16 __attribute__((ext_vector_type(16)));
typedef _Float16 half8  __attribute__((ext_vector_type(8)));
typedef float    float8 __attribute__((ext_vector_type(8)));

#define NN 100000   // nodes
#define NE 800000   // edges
#define NG 512      // graphs

// monotone float<->uint for atomic min/max
__device__ __forceinline__ unsigned fenc(float f) {
  unsigned u = __float_as_uint(f);
  return (u & 0x80000000u) ? ~u : (u | 0x80000000u);
}
__device__ __forceinline__ float fdec(unsigned k) {
  unsigned u = (k & 0x80000000u) ? (k ^ 0x80000000u) : ~k;
  return __uint_as_float(u);
}

// CDNA5 ISA 7.12.2 fragment k-striping
__device__ __forceinline__ int ka_idx(int i, int hi) { return i + (i & 8) + (hi << 3); } // A 16x32
__device__ __forceinline__ int kb_idx(int i, int hi) { return i + (i & 8) + (hi << 3); } // B 32x16

// A fragment = two aligned 16B loads (halves 0..7 and 16..23 of the 32-half row
// window give exactly the ISA striping when offset by hi*8)
__device__ __forceinline__ half16 ld_frag(const _Float16* p) {
  half8 lo = *(const half8*)p;
  half8 hv = *(const half8*)(p + 16);
  half16 r;
#pragma unroll
  for (int i = 0; i < 8; ++i) { r[i] = lo[i]; r[i + 8] = hv[i]; }
  return r;
}
// packed-B fragment: 16 contiguous halfs per lane
__device__ __forceinline__ half16 ld_fragB(const _Float16* p) {
  half8 lo = *(const half8*)p;
  half8 hv = *(const half8*)(p + 8);
  half16 r;
#pragma unroll
  for (int i = 0; i < 8; ++i) { r[i] = lo[i]; r[i + 8] = hv[i]; }
  return r;
}

// ---------------------------------------------------------------------------
__global__ void k_fill_u32(unsigned* p, unsigned v, long n) {
  long i = (long)blockIdx.x * blockDim.x + threadIdx.x;
  long st = (long)gridDim.x * blockDim.x;
  for (; i < n; i += st) p[i] = v;
}

__global__ void k_degree(const int* dst, float* deg, int E) {
  long i = (long)blockIdx.x * blockDim.x + threadIdx.x;
  long st = (long)gridDim.x * blockDim.x;
  for (; i < E; i += st) atomicAdd(&deg[dst[i]], 1.0f);
}

__global__ void k_delta(const float* deg, float* accum, int n) {
  __shared__ float red[256];
  float s = 0.f;
  for (long i = (long)blockIdx.x * blockDim.x + threadIdx.x; i < n;
       i += (long)gridDim.x * blockDim.x)
    s += log1pf(deg[i]);
  red[threadIdx.x] = s;
  __syncthreads();
  for (int w = 128; w > 0; w >>= 1) {
    if ((int)threadIdx.x < w) red[threadIdx.x] += red[threadIdx.x + w];
    __syncthreads();
  }
  if (threadIdx.x == 0) atomicAdd(accum, red[0]);
}

__global__ void k_scalers(const float* deg, const float* accum, float* samp, float* satt, int n) {
  int i = blockIdx.x * blockDim.x + threadIdx.x;
  if (i >= n) return;
  float delta = accum[0] / (float)n;
  float ld = log1pf(fmaxf(deg[i], 1.0f));
  samp[i] = ld / delta;
  satt[i] = delta / ld;
}

__global__ void k_f32_to_f16(const float* in, _Float16* out, long n) {
  long i = (long)blockIdx.x * blockDim.x + threadIdx.x;
  long st = (long)gridDim.x * blockDim.x;
  for (; i < n; i += st) out[i] = (_Float16)in[i];
}

// ---- weight prep: pack B matrices in fragment order -----------------------
// edge B: [tile(7)][chunk(4)][lane(32)][i(16)], K=100 pad 128, N=100 pad 112
__global__ void k_prep_edge_w(const float* pre_w, _Float16* Bp) {
  int id = blockIdx.x * blockDim.x + threadIdx.x;     // 7*4*32*16 = 14336
  if (id >= 7 * 4 * 32 * 16) return;
  int i = id & 15, lane = (id >> 4) & 31, c = (id >> 9) & 3, tile = id >> 11;
  int k = c * 32 + kb_idx(i, lane >> 4);
  int n = tile * 16 + (lane & 15);
  float v = 0.f;
  if (k < 100 && n < 100) v = pre_w[(n / 50) * 5000 + k * 50 + (n % 50)];
  Bp[id] = (_Float16)v;
}

// post B: [t(2)][chunk(CH)][tn(2)][lane(32)][i(16)], rows permuted to
// [agg(12f) | x(f)] to match the LDS-staged A layout.
__global__ void k_prep_post_w(const float* post_w, _Float16* Pp, int f, int CH) {
  long tot = 2l * CH * 2 * 32 * 16;
  long id = (long)blockIdx.x * blockDim.x + threadIdx.x;
  long st = (long)gridDim.x * blockDim.x;
  int f12 = 12 * f, f13 = 13 * f;
  for (; id < tot; id += st) {
    int i = (int)(id & 15);
    int lane = (int)((id >> 4) & 31);
    int tn = (int)((id >> 9) & 1);
    long r = id >> 10;                // t*CH + c
    int c = (int)(r % CH);
    int t = (int)(r / CH);
    int p = c * 32 + kb_idx(i, lane >> 4);   // padded/permuted row
    int n = tn * 16 + (lane & 15);
    float v = 0.f;
    if (p < f13 && n < 25) {
      int kref = (p < f12) ? (p + f) : (p - f12);   // un-permute
      v = post_w[t * f13 * 25 + kref * 25 + n];
    }
    Pp[id] = (_Float16)v;
  }
}

// lin B: [tile(4)][chunk(2)][lane(32)][i(16)], 50x50 pad 64x64
__global__ void k_prep_lin_w(const float* lin_w, _Float16* Lp) {
  int id = blockIdx.x * blockDim.x + threadIdx.x;     // 4*2*32*16 = 4096
  if (id >= 4 * 2 * 32 * 16) return;
  int i = id & 15, lane = (id >> 4) & 31, c = (id >> 9) & 1, tile = id >> 10;
  int k = c * 32 + kb_idx(i, lane >> 4);
  int n = tile * 16 + (lane & 15);
  float v = (k < 50 && n < 50) ? lin_w[k * 50 + n] : 0.f;
  Lp[id] = (_Float16)v;
}

// ---- layer 0 edge path (f=1, scalar) --------------------------------------
__global__ void k_edge_f1(const float* x, const int* src, const int* dst,
                          const float* pre_w, const float* pre_b,
                          float* sum, float* sum2, unsigned* mn, unsigned* mx, int E) {
  long i = (long)blockIdx.x * blockDim.x + threadIdx.x;
  long st = (long)gridDim.x * blockDim.x;
  for (; i < 2l * E; i += st) {
    int e = (int)(i >> 1), t = (int)(i & 1);
    int d = dst[e], s = src[e];
    float m = x[d] * pre_w[t * 2 + 0] + x[s] * pre_w[t * 2 + 1] + pre_b[t];
    int idx = d * 2 + t;
    atomicAdd(&sum[idx], m);
    atomicAdd(&sum2[idx], m * m);
    atomicMin(&mn[idx], fenc(m));
    atomicMax(&mx[idx], fenc(m));
  }
}

// ---- edge GEMM + scatter, f=50 layers (WMMA) ------------------------------
// 16 edges/wave; K=100 pad 128; N=100 (both towers) pad 112.
__global__ __launch_bounds__(32)
void k_edge_wmma(const _Float16* xh, const int* src, const int* dst,
                 const _Float16* Bp, const float* pre_b,
                 float* sum, float* sum2, unsigned* mn, unsigned* mx) {
  __shared__ _Float16 hbuf[16 * 128];          // [row][dst(50)|src(50)|pad(28)]
  const int base = blockIdx.x * 16;
  const int lane = threadIdx.x;
  const int row = lane & 15, hi = lane >> 4, col = lane & 15;

  __builtin_prefetch(dst + base + 16, 0, 3);   // global_prefetch_b8
  __builtin_prefetch(src + base + 16, 0, 3);

  // stage h rows: lane (row,hi) copies 50 halfs of x[dst] (hi=0) / x[src] (hi=1)
  {
    int e = base + row;
    int node = (hi == 0) ? dst[e] : src[e];
    const unsigned* s32 = (const unsigned*)(xh + node * 50);
    unsigned* d32 = (unsigned*)(&hbuf[row * 128 + hi * 50]);
#pragma unroll
    for (int q = 0; q < 25; ++q) d32[q] = s32[q];
    if (hi == 0) {                              // zero pad cols 100..127
      unsigned* z = (unsigned*)(&hbuf[row * 128 + 100]);
#pragma unroll
      for (int q = 0; q < 14; ++q) z[q] = 0u;
    }
  }
  __syncthreads();

  // A fragments from LDS (ds_load_b128 pairs)
  half16 a0 = ld_frag(&hbuf[row * 128 + 0 * 32 + hi * 8]);
  half16 a1 = ld_frag(&hbuf[row * 128 + 1 * 32 + hi * 8]);
  half16 a2 = ld_frag(&hbuf[row * 128 + 2 * 32 + hi * 8]);
  half16 a3 = ld_frag(&hbuf[row * 128 + 3 * 32 + hi * 8]);

  int drow[8];
#pragma unroll
  for (int r = 0; r < 8; ++r) drow[r] = dst[base + (hi << 3) + r];

#pragma unroll
  for (int tile = 0; tile < 7; ++tile) {
    float8 acc = {};
    const _Float16* bp = Bp + (size_t)(tile * 4) * 32 * 16 + lane * 16;
    acc = __builtin_amdgcn_wmma_f32_16x16x32_f16(false, a0, false, ld_fragB(bp + 0 * 512),
                                                 (short)0, acc, false, false);
    acc = __builtin_amdgcn_wmma_f32_16x16x32_f16(false, a1, false, ld_fragB(bp + 1 * 512),
                                                 (short)0, acc, false, false);
    acc = __builtin_amdgcn_wmma_f32_16x16x32_f16(false, a2, false, ld_fragB(bp + 2 * 512),
                                                 (short)0, acc, false, false);
    acc = __builtin_amdgcn_wmma_f32_16x16x32_f16(false, a3, false, ld_fragB(bp + 3 * 512),
                                                 (short)0, acc, false, false);
    int nch = tile * 16 + col;
    if (nch < 100) {
      float bias = pre_b[nch];                 // pre_b flat (T,f) == t*50+g
#pragma unroll
      for (int r = 0; r < 8; ++r) {
        float v = acc[r] + bias;
        int idx = drow[r] * 100 + nch;
        atomicAdd(&sum[idx], v);
        atomicAdd(&sum2[idx], v * v);
        atomicMin(&mn[idx], fenc(v));
        atomicMax(&mx[idx], fenc(v));
      }
    }
  }
}

// ---- finalize aggregation -> aggH (f16, [n][t][{mean,mn,mx,std}*f]) -------
__global__ void k_finalize_agg(const float* deg, const float* sum, const float* sum2,
                               const unsigned* mn, const unsigned* mx,
                               _Float16* aggH, int n_nodes, int f) {
  int TF = 2 * f;
  long tot = (long)n_nodes * TF;
  long i = (long)blockIdx.x * blockDim.x + threadIdx.x;
  long st = (long)gridDim.x * blockDim.x;
  for (; i < tot; i += st) {
    int n = (int)(i / TF), tf = (int)(i % TF);
    int t = tf / f, g = tf % f;
    float dc = fmaxf(deg[n], 1.0f);
    float me = sum[i] / dc;
    float m2 = sum2[i] / dc;
    float sd = sqrtf(fmaxf(m2 - me * me, 0.f) + 1e-5f);
    unsigned kmn = mn[i], kmx = mx[i];
    float vmn = (kmn == 0xFFFFFFFFu) ? 0.f : fdec(kmn);
    float vmx = (kmx == 0u) ? 0.f : fdec(kmx);
    int bofs = n * 8 * f + t * 4 * f;
    aggH[bofs + g]         = (_Float16)me;
    aggH[bofs + f + g]     = (_Float16)vmn;
    aggH[bofs + 2 * f + g] = (_Float16)vmx;
    aggH[bofs + 3 * f + g] = (_Float16)sd;
  }
}

// ---- node: post GEMM (per tower) + lin GEMM, all WMMA ---------------------
// LDS row layout (permuted, matches Pp): [agg*scales(12F) | x(F) | pad]
template <int F>
__global__ __launch_bounds__(32)
void k_node_wmma(const _Float16* xh, const _Float16* aggH,
                 const float* samp, const float* satt,
                 const _Float16* Pp, const float* post_b,
                 const _Float16* Lp, const float* lin_b,
                 float* y_out, float* bn_sum, float* bn_sum2) {
  constexpr int CH = (13 * F + 31) / 32;
  constexpr int W = CH * 32;                  // padded K
  constexpr int F4 = 4 * F, F8 = 8 * F, F12 = 12 * F, F13 = 13 * F;
  __shared__ _Float16 ob[16 * W];
  __shared__ _Float16 yh[16 * 64];

  const int lane = threadIdx.x;
  const int row = lane & 15, hi = lane >> 4, col = lane & 15;
  const int base = blockIdx.x * 16;

  for (int i = lane; i < 16 * 64; i += 32) yh[i] = (_Float16)0;
  // stage x part (shared by towers) + zero pad
  for (int idx = lane; idx < 16 * F; idx += 32) {
    int r = idx / F, k = idx - r * F;
    ob[r * W + F12 + k] = xh[(base + r) * F + k];
  }
  for (int idx = lane; idx < 16 * (W - F13); idx += 32) {
    int r = idx / (W - F13), q = idx - r * (W - F13);
    ob[r * W + F13 + q] = (_Float16)0;
  }
  __syncthreads();

#pragma unroll
  for (int t = 0; t < 2; ++t) {
    // stage agg*{1, s_amp, s_att}: one load feeds all three scale blocks
    if constexpr (F4 % 8 == 0) {
      constexpr int Q = F4 / 8;
      for (int idx = lane; idx < 16 * Q; idx += 32) {
        int r = idx / Q, q = idx - r * Q;
        int n2 = base + r;
        half8 v = *(const half8*)&aggH[n2 * F8 + t * F4 + q * 8];
        float sa2 = samp[n2], st2 = satt[n2];
        half8 v1, v2;
#pragma unroll
        for (int i = 0; i < 8; ++i) {
          float fv = (float)v[i];
          v1[i] = (_Float16)(fv * sa2);
          v2[i] = (_Float16)(fv * st2);
        }
        *(half8*)&ob[r * W + q * 8]            = v;
        *(half8*)&ob[r * W + F4 + q * 8]       = v1;
        *(half8*)&ob[r * W + 2 * F4 + q * 8]   = v2;
      }
    } else {
      for (int idx = lane; idx < 16 * F12; idx += 32) {
        int r = idx / F12, j = idx - r * F12;
        int blk = j / F4, q = j - blk * F4;
        int n2 = base + r;
        float fv = (float)aggH[n2 * F8 + t * F4 + q];
        float sc = (blk == 0) ? 1.f : ((blk == 1) ? samp[n2] : satt[n2]);
        ob[r * W + j] = (_Float16)(fv * sc);
      }
    }
    __syncthreads();

    float8 acc0 = {}, acc1 = {};
#pragma unroll
    for (int c = 0; c < CH; ++c) {
      half16 af = ld_frag(&ob[row * W + c * 32 + hi * 8]);
      const _Float16* pp = Pp + ((size_t)(t * CH + c) * 2) * 32 * 16 + lane * 16;
      acc0 = __builtin_amdgcn_wmma_f32_16x16x32_f16(false, af, false, ld_fragB(pp),
                                                    (short)0, acc0, false, false);
      acc1 = __builtin_amdgcn_wmma_f32_16x16x32_f16(false, af, false, ld_fragB(pp + 512),
                                                    (short)0, acc1, false, false);
    }
#pragma unroll
    for (int r = 0; r < 8; ++r) {
      int m = r + (hi << 3);
      int ch0 = col;                           // 0..15
      yh[m * 64 + t * 25 + ch0] = (_Float16)(acc0[r] + post_b[t * 25 + ch0]);
      int ch1 = 16 + col;                      // 16..31
      if (ch1 < 25)
        yh[m * 64 + t * 25 + ch1] = (_Float16)(acc1[r] + post_b[t * 25 + ch1]);
    }
    __syncthreads();
  }

  // lin: (16x50 pad64) x (50x50 pad 64x64)
#pragma unroll
  for (int tile = 0; tile < 4; ++tile) {
    float8 acc = {};
#pragma unroll
    for (int c = 0; c < 2; ++c) {
      half16 af = ld_frag(&yh[row * 64 + c * 32 + hi * 8]);
      const _Float16* lp = Lp + (size_t)(tile * 2 + c) * 32 * 16 + lane * 16;
      acc = __builtin_amdgcn_wmma_f32_16x16x32_f16(false, af, false, ld_fragB(lp),
                                                   (short)0, acc, false, false);
    }
    int ch = tile * 16 + col;
    if (ch < 50) {
      float lb = lin_b[ch];
      float s1 = 0.f, s2 = 0.f;
#pragma unroll
      for (int r = 0; r < 8; ++r) {
        float v = acc[r] + lb;
        y_out[(base + r + (hi << 3)) * 50 + ch] = v;
        s1 += v;
        s2 += v * v;
      }
      atomicAdd(&bn_sum[ch], s1);
      atomicAdd(&bn_sum2[ch], s2);
    }
  }
}

// ---- batch norm -----------------------------------------------------------
__global__ void k_bn_stats(const float* bn_sum, const float* bn_sum2,
                           const float* gamma, const float* beta,
                           float* scale, float* shift, int n) {
  int c = threadIdx.x;
  if (c < 50) {
    float mu = bn_sum[c] / (float)n;
    float var = bn_sum2[c] / (float)n - mu * mu;
    float sc = gamma[c] * rsqrtf(var + 1e-5f);
    scale[c] = sc;
    shift[c] = beta[c] - mu * sc;
  }
}

__global__ void k_bn_apply(float* x, _Float16* xh, const float* scale,
                           const float* shift, long total) {
  long i = (long)blockIdx.x * blockDim.x + threadIdx.x;
  long st = (long)gridDim.x * blockDim.x;
  for (; i < total; i += st) {
    int c = (int)(i % 50);
    float v = fmaxf(x[i] * scale[c] + shift[c], 0.f);
    x[i] = v;
    xh[i] = (_Float16)v;
  }
}

// ---- pooling + MLP --------------------------------------------------------
__global__ void k_pool(const float* x, const int* batch, float* g, long total) {
  long i = (long)blockIdx.x * blockDim.x + threadIdx.x;
  long st = (long)gridDim.x * blockDim.x;
  for (; i < total; i += st) {
    int n = (int)(i / 50), c = (int)(i % 50);
    atomicAdd(&g[batch[n] * 50 + c], x[i]);
  }
}

__global__ __launch_bounds__(64)
void k_mlp(const float* g, const float* w1, const float* b1,
           const float* w2, const float* b2,
           const float* w3, const float* b3, float* out) {
  __shared__ float gi[64], h1[64], h2[32];
  int j = threadIdx.x, blk = blockIdx.x;
  if (j < 50) gi[j] = g[blk * 50 + j];
  __syncthreads();
  if (j < 50) {
    float a = b1[j];
#pragma unroll
    for (int k = 0; k < 50; ++k) a += gi[k] * w1[k * 50 + j];
    h1[j] = fmaxf(a, 0.f);
  }
  __syncthreads();
  if (j < 25) {
    float a = b2[j];
#pragma unroll
    for (int k = 0; k < 50; ++k) a += h1[k] * w2[k * 25 + j];
    h2[j] = fmaxf(a, 0.f);
  }
  __syncthreads();
  if (j == 0) {
    float a = b3[0];
#pragma unroll
    for (int k = 0; k < 25; ++k) a += h2[k] * w3[k];
    out[blk] = a;
  }
}

// ---------------------------------------------------------------------------
static inline void fillu(unsigned* p, unsigned v, long n, hipStream_t s) {
  long b = (n + 255) / 256;
  if (b > 2048) b = 2048;
  if (b < 1) b = 1;
  k_fill_u32<<<dim3((unsigned)b), dim3(256), 0, s>>>(p, v, n);
}

extern "C" void kernel_launch(void* const* d_in, const int* in_sizes, int n_in,
                              void* d_out, int out_size, void* d_ws, size_t ws_size,
                              hipStream_t stream) {
  (void)in_sizes; (void)n_in; (void)out_size; (void)ws_size;
  const int N = NN, E = NE, G = NG;

  // inputs in setup_inputs() insertion order, params depth-first:
  // 0:x 1:edge_index 2:batch, then per layer (pre_w,pre_b,post_w,post_b,
  // lin_w,lin_b,bn_gamma,bn_beta) x4, then mlp w1,b1,w2,b2,w3,b3
  const float* x_in = (const float*)d_in[0];
  const int* ei = (const int*)d_in[1];
  const int* srcp = ei;
  const int* dstp = ei + E;
  const int* batch = (const int*)d_in[2];
  const float* mw1 = (const float*)d_in[35];
  const float* mb1 = (const float*)d_in[36];
  const float* mw2 = (const float*)d_in[37];
  const float* mb2 = (const float*)d_in[38];
  const float* mw3 = (const float*)d_in[39];
  const float* mb3 = (const float*)d_in[40];
  float* out = (float*)d_out;

  char* w = (char*)d_ws;
  auto carve = [&](size_t bytes) -> char* {
    char* p = w;
    w += (bytes + 255) & ~(size_t)255;
    return p;
  };
  float* deg      = (float*)carve((size_t)N * 4);
  float* scal     = (float*)carve(256);
  float* sampv    = (float*)carve((size_t)N * 4);
  float* sattv    = (float*)carve((size_t)N * 4);
  float* x_out    = (float*)carve((size_t)N * 50 * 4);
  _Float16* x_h   = (_Float16*)carve((size_t)N * 50 * 2);
  float* sumB     = (float*)carve((size_t)N * 100 * 4);
  float* sum2B    = (float*)carve((size_t)N * 100 * 4);
  unsigned* mnB   = (unsigned*)carve((size_t)N * 100 * 4);
  unsigned* mxB   = (unsigned*)carve((size_t)N * 100 * 4);
  _Float16* aggH  = (_Float16*)carve((size_t)N * 400 * 2);
  _Float16* BpE   = (_Float16*)carve((size_t)7 * 4 * 32 * 16 * 2);
  _Float16* Pp    = (_Float16*)carve((size_t)2 * 21 * 2 * 32 * 16 * 2);
  _Float16* Lp    = (_Float16*)carve((size_t)4 * 2 * 32 * 16 * 2);
  float* bn_sum   = (float*)carve(64 * 4);
  float* bn_sum2  = (float*)carve(64 * 4);
  float* bn_scale = (float*)carve(64 * 4);
  float* bn_shift = (float*)carve(64 * 4);
  float* gbuf     = (float*)carve((size_t)G * 50 * 4);

  // degree + scalers
  fillu((unsigned*)deg, 0u, N, stream);
  fillu((unsigned*)scal, 0u, 64, stream);
  k_degree<<<2048, 256, 0, stream>>>(dstp, deg, E);
  k_delta<<<256, 256, 0, stream>>>(deg, scal, N);
  k_scalers<<<(N + 255) / 256, 256, 0, stream>>>(deg, scal, sampv, sattv, N);
  k_f32_to_f16<<<(N + 255) / 256, 256, 0, stream>>>(x_in, x_h, N);

  for (int L = 0; L < 4; ++L) {
    const int f = (L == 0) ? 1 : 50;
    const int TF = 2 * f;
    const int chunks = (13 * f + 31) / 32;
    const float* pre_w  = (const float*)d_in[3 + L * 8 + 0];
    const float* pre_b  = (const float*)d_in[3 + L * 8 + 1];
    const float* post_w = (const float*)d_in[3 + L * 8 + 2];
    const float* post_b = (const float*)d_in[3 + L * 8 + 3];
    const float* lin_w  = (const float*)d_in[3 + L * 8 + 4];
    const float* lin_b  = (const float*)d_in[3 + L * 8 + 5];
    const float* gamma  = (const float*)d_in[3 + L * 8 + 6];
    const float* beta   = (const float*)d_in[3 + L * 8 + 7];

    fillu((unsigned*)sumB, 0u, (long)N * TF, stream);
    fillu((unsigned*)sum2B, 0u, (long)N * TF, stream);
    fillu(mnB, 0xFFFFFFFFu, (long)N * TF, stream);
    fillu(mxB, 0u, (long)N * TF, stream);
    fillu((unsigned*)bn_sum, 0u, 64, stream);
    fillu((unsigned*)bn_sum2, 0u, 64, stream);

    if (L > 0) k_prep_edge_w<<<(14336 + 255) / 256, 256, 0, stream>>>(pre_w, BpE);
    k_prep_post_w<<<(2 * chunks * 1024 + 255) / 256, 256, 0, stream>>>(post_w, Pp, f, chunks);
    k_prep_lin_w<<<(4096 + 255) / 256, 256, 0, stream>>>(lin_w, Lp);

    if (L == 0)
      k_edge_f1<<<2048, 256, 0, stream>>>(x_in, srcp, dstp, pre_w, pre_b,
                                          sumB, sum2B, mnB, mxB, E);
    else
      k_edge_wmma<<<E / 16, 32, 0, stream>>>(x_h, srcp, dstp, BpE, pre_b,
                                             sumB, sum2B, mnB, mxB);

    k_finalize_agg<<<2048, 256, 0, stream>>>(deg, sumB, sum2B, mnB, mxB, aggH, N, f);

    if (L == 0)
      k_node_wmma<1><<<N / 16, 32, 0, stream>>>(x_h, aggH, sampv, sattv, Pp,
                                                post_b, Lp, lin_b, x_out,
                                                bn_sum, bn_sum2);
    else
      k_node_wmma<50><<<N / 16, 32, 0, stream>>>(x_h, aggH, sampv, sattv, Pp,
                                                 post_b, Lp, lin_b, x_out,
                                                 bn_sum, bn_sum2);

    k_bn_stats<<<1, 64, 0, stream>>>(bn_sum, bn_sum2, gamma, beta, bn_scale, bn_shift, N);
    k_bn_apply<<<2048, 256, 0, stream>>>(x_out, x_h, bn_scale, bn_shift, (long)N * 50);
  }

  fillu((unsigned*)gbuf, 0u, (long)G * 50, stream);
  k_pool<<<2048, 256, 0, stream>>>(x_out, batch, gbuf, (long)N * 50);
  k_mlp<<<G, 64, 0, stream>>>(gbuf, mw1, mb1, mw2, mb2, mw3, mb3, out);
}
